// BTGINs_88098369176165
// MI455X (gfx1250) — matrix operand order repaired and checked by reading, not compiled, not run on hardware
//
#include <hip/hip_runtime.h>
#include <hip/hip_bf16.h>

#define NNODES 100000
#define NEDGES 1600000
#define FDIM   128
#define NPAIR  (NNODES / 32)          // 3125 row-pairs of 32 rows
#define NF     ((size_t)NNODES * FDIM)

typedef __bf16 v16bf __attribute__((ext_vector_type(16)));
typedef float  v8f   __attribute__((ext_vector_type(8)));

__device__ __forceinline__ void atomic_add_f32(float* p, float v) {
    unsafeAtomicAdd(p, v);            // native global_atomic_add_f32
}

// ---------------------------------------------------------------- zero
__global__ void zero_f4_kernel(float4* __restrict__ p, int n4) {
    int i = blockIdx.x * 256 + threadIdx.x;
    if (i < n4) p[i] = make_float4(0.f, 0.f, 0.f, 0.f);
}

// ---------------------------------------------------------------- weight pack
// Pack W (128x128 row-major f32, W[k*128+n]) into per-lane B fragments:
// P[(((t*4)+kit)*32 + lane)*16 + i] = bf16(W[(kit*32 + (lane>>4)*8 + (i&7) + (i>>3)*16)*128 + t*16 + (lane&15)])
__global__ void pack_w_kernel(const float* __restrict__ W, __bf16* __restrict__ P) {
    int idx  = blockIdx.x * 256 + threadIdx.x;   // 16384 total
    int i    = idx & 15;
    int lane = (idx >> 4) & 31;
    int kit  = (idx >> 9) & 3;
    int t    = (idx >> 11) & 7;
    int k    = kit * 32 + ((lane >> 4) * 8) + (i & 7) + ((i >> 3) * 16);
    int n    = t * 16 + (lane & 15);
    P[idx] = (__bf16)W[k * FDIM + n];
}

// ---------------------------------------------------------------- edge scatter
// 32 threads per edge, float4 per thread; x[src] row gathered, atomically
// summed into agg[dst] row. Working set is L2-resident (51.2 MB).
__global__ void scatter_kernel(const float* __restrict__ X,
                               const int* __restrict__ src,
                               const int* __restrict__ dst,
                               float* __restrict__ AGG) {
    int idx = blockIdx.x * 256 + threadIdx.x;
    int e   = idx >> 5;
    if (e >= NEDGES) return;
    int f = (idx & 31) * 4;
    int s = src[e], d = dst[e];
    const float4 v = *reinterpret_cast<const float4*>(X + s * FDIM + f);
    float* o = AGG + d * FDIM + f;
    atomic_add_f32(o + 0, v.x);
    atomic_add_f32(o + 1, v.y);
    atomic_add_f32(o + 2, v.z);
    atomic_add_f32(o + 3, v.w);
}

// ---------------------------------------------------------------- BN finalize
__global__ void bn_finalize_kernel(const float* __restrict__ SUM,
                                   const float* __restrict__ SSQ,
                                   const float* __restrict__ g,
                                   const float* __restrict__ bt,
                                   float* __restrict__ SC,
                                   float* __restrict__ SH) {
    int c = threadIdx.x;                       // 128 threads
    const float invN = 1.0f / (float)NNODES;
    float mu  = SUM[c] * invN;
    float var = SSQ[c] * invN - mu * mu;       // biased variance (torch BN training)
    float s   = g[c] * rsqrtf(var + 1e-5f);
    SC[c] = s;
    SH[c] = bt[c] - mu * s;
}

// A-fragment builders (ISA 16-bit A layout: lane<16 -> K {0..7,16..23},
// lane>=16 -> K {8..15,24..31} relative to K0). Explicit float4 loads ->
// guaranteed global_load_b128 with 32-bit saddr+voffset addressing.
__device__ __forceinline__ v16bf build_a_gin(const float* __restrict__ X,
                                             const float* __restrict__ G,
                                             int rb, int K0, int kb, float ep) {
    v16bf a;
#pragma unroll
    for (int j = 0; j < 2; ++j) {
        const int o = rb + K0 + kb + j * 16;
        const float4 x0 = *reinterpret_cast<const float4*>(X + o);
        const float4 x1 = *reinterpret_cast<const float4*>(X + o + 4);
        const float4 g0 = *reinterpret_cast<const float4*>(G + o);
        const float4 g1 = *reinterpret_cast<const float4*>(G + o + 4);
        a[j * 8 + 0] = (__bf16)__builtin_fmaf(ep, x0.x, g0.x);
        a[j * 8 + 1] = (__bf16)__builtin_fmaf(ep, x0.y, g0.y);
        a[j * 8 + 2] = (__bf16)__builtin_fmaf(ep, x0.z, g0.z);
        a[j * 8 + 3] = (__bf16)__builtin_fmaf(ep, x0.w, g0.w);
        a[j * 8 + 4] = (__bf16)__builtin_fmaf(ep, x1.x, g1.x);
        a[j * 8 + 5] = (__bf16)__builtin_fmaf(ep, x1.y, g1.y);
        a[j * 8 + 6] = (__bf16)__builtin_fmaf(ep, x1.z, g1.z);
        a[j * 8 + 7] = (__bf16)__builtin_fmaf(ep, x1.w, g1.w);
    }
    return a;
}

__device__ __forceinline__ v16bf build_a_bn(const float* __restrict__ Z,
                                            const float* __restrict__ SC,
                                            const float* __restrict__ SH,
                                            int rb, int K0, int kb) {
    v16bf a;
#pragma unroll
    for (int j = 0; j < 2; ++j) {
        const int ko = K0 + kb + j * 16;
        const int o  = rb + ko;
        const float4 z0 = *reinterpret_cast<const float4*>(Z + o);
        const float4 z1 = *reinterpret_cast<const float4*>(Z + o + 4);
        const float4 s0 = *reinterpret_cast<const float4*>(SC + ko);
        const float4 s1 = *reinterpret_cast<const float4*>(SC + ko + 4);
        const float4 h0 = *reinterpret_cast<const float4*>(SH + ko);
        const float4 h1 = *reinterpret_cast<const float4*>(SH + ko + 4);
        a[j * 8 + 0] = (__bf16)fmaxf(__builtin_fmaf(z0.x, s0.x, h0.x), 0.f);
        a[j * 8 + 1] = (__bf16)fmaxf(__builtin_fmaf(z0.y, s0.y, h0.y), 0.f);
        a[j * 8 + 2] = (__bf16)fmaxf(__builtin_fmaf(z0.z, s0.z, h0.z), 0.f);
        a[j * 8 + 3] = (__bf16)fmaxf(__builtin_fmaf(z0.w, s0.w, h0.w), 0.f);
        a[j * 8 + 4] = (__bf16)fmaxf(__builtin_fmaf(z1.x, s1.x, h1.x), 0.f);
        a[j * 8 + 5] = (__bf16)fmaxf(__builtin_fmaf(z1.y, s1.y, h1.y), 0.f);
        a[j * 8 + 6] = (__bf16)fmaxf(__builtin_fmaf(z1.z, s1.z, h1.z), 0.f);
        a[j * 8 + 7] = (__bf16)fmaxf(__builtin_fmaf(z1.w, s1.w, h1.w), 0.f);
    }
    return a;
}

// ---------------------------------------------------------------- GEMM-a
// Z = ((1+eps)*Xin + AGG) @ W + bias, plus per-column sum/sumsq for BN.
// One wave -> 32 rows x 128 cols: 2 A-frags x 8 B-tiles, 16 wmma per k-step.
__global__ void __launch_bounds__(256)
gemm_a_kernel(const float* __restrict__ Xin, const float* __restrict__ AGG,
              const float* __restrict__ eps_ptr,
              const __bf16* __restrict__ BP, const float* __restrict__ bias,
              float* __restrict__ Z, float* __restrict__ SUM, float* __restrict__ SSQ) {
    const int lane = threadIdx.x & 31;
    const int wave = threadIdx.x >> 5;
    const int pair = blockIdx.x * 8 + wave;
    if (pair >= NPAIR) return;
    const int m0   = pair * 32;
    const int ncol = lane & 15;
    const int lh   = lane >> 4;
    const int kb   = lh * 8;
    const int rb0  = (m0 + ncol) * FDIM;     // 32-bit offsets (<= 12.8M elements)
    const int rb1  = rb0 + 16 * FDIM;
    const float ep = 1.0f + eps_ptr[0];

    v8f acc[2][8];
#pragma unroll
    for (int t = 0; t < 8; ++t) {
        float bv = bias[t * 16 + ncol];
#pragma unroll
        for (int r = 0; r < 8; ++r) { acc[0][t][r] = bv; acc[1][t][r] = bv; }
    }

#pragma unroll
    for (int kit = 0; kit < 4; ++kit) {
        const int K0 = kit * 32;
        v16bf a0 = build_a_gin(Xin, AGG, rb0, K0, kb, ep);
        v16bf a1 = build_a_gin(Xin, AGG, rb1, K0, kb, ep);
        v16bf b[8];
#pragma unroll
        for (int t = 0; t < 8; ++t)
            b[t] = *reinterpret_cast<const v16bf*>(BP + (((t * 4 + kit) * 32 + lane) << 4));
#pragma unroll
        for (int t = 0; t < 8; ++t) {
            acc[0][t] = __builtin_amdgcn_wmma_f32_16x16x32_bf16(
                false, a0, false, b[t], (short)0, acc[0][t], false, false);
            acc[1][t] = __builtin_amdgcn_wmma_f32_16x16x32_bf16(
                false, a1, false, b[t], (short)0, acc[1][t], false, false);
        }
    }

#pragma unroll
    for (int t = 0; t < 8; ++t) {
        const int c = t * 16 + ncol;
        float s = 0.f, ss = 0.f;
#pragma unroll
        for (int h = 0; h < 2; ++h) {
#pragma unroll
            for (int r = 0; r < 8; ++r) {
                float v = acc[h][t][r];
                Z[(m0 + h * 16 + r + lh * 8) * FDIM + c] = v;
                s += v; ss += v * v;
            }
        }
        s  += __shfl_xor(s, 16, 32);           // combine the two lane-halves (same column)
        ss += __shfl_xor(ss, 16, 32);
        if (lh == 0) {
            atomic_add_f32(&SUM[c], s);
            atomic_add_f32(&SSQ[c], ss);
        }
    }
}

// ---------------------------------------------------------------- GEMM-b
// Out = relu?(relu(Z*SC+SH) @ W + bias). BN+ReLU fused into A-fragment build.
__global__ void __launch_bounds__(256)
gemm_b_kernel(const float* __restrict__ Zin,
              const float* __restrict__ SC, const float* __restrict__ SH,
              const __bf16* __restrict__ BP, const float* __restrict__ bias,
              float* __restrict__ Out, const int relu_out) {
    const int lane = threadIdx.x & 31;
    const int wave = threadIdx.x >> 5;
    const int pair = blockIdx.x * 8 + wave;
    if (pair >= NPAIR) return;
    const int m0   = pair * 32;
    const int ncol = lane & 15;
    const int lh   = lane >> 4;
    const int kb   = lh * 8;
    const int rb0  = (m0 + ncol) * FDIM;
    const int rb1  = rb0 + 16 * FDIM;

    v8f acc[2][8];
#pragma unroll
    for (int t = 0; t < 8; ++t) {
        float bv = bias[t * 16 + ncol];
#pragma unroll
        for (int r = 0; r < 8; ++r) { acc[0][t][r] = bv; acc[1][t][r] = bv; }
    }

#pragma unroll
    for (int kit = 0; kit < 4; ++kit) {
        const int K0 = kit * 32;
        v16bf a0 = build_a_bn(Zin, SC, SH, rb0, K0, kb);
        v16bf a1 = build_a_bn(Zin, SC, SH, rb1, K0, kb);
        v16bf b[8];
#pragma unroll
        for (int t = 0; t < 8; ++t)
            b[t] = *reinterpret_cast<const v16bf*>(BP + (((t * 4 + kit) * 32 + lane) << 4));
#pragma unroll
        for (int t = 0; t < 8; ++t) {
            acc[0][t] = __builtin_amdgcn_wmma_f32_16x16x32_bf16(
                false, a0, false, b[t], (short)0, acc[0][t], false, false);
            acc[1][t] = __builtin_amdgcn_wmma_f32_16x16x32_bf16(
                false, a1, false, b[t], (short)0, acc[1][t], false, false);
        }
    }

#pragma unroll
    for (int t = 0; t < 8; ++t) {
        const int c = t * 16 + ncol;
#pragma unroll
        for (int h = 0; h < 2; ++h) {
#pragma unroll
            for (int r = 0; r < 8; ++r) {
                float v = acc[h][t][r];
                if (relu_out) v = fmaxf(v, 0.f);
                Out[(m0 + h * 16 + r + lh * 8) * FDIM + c] = v;
            }
        }
    }
}

// ---------------------------------------------------------------- launcher
extern "C" void kernel_launch(void* const* d_in, const int* in_sizes, int n_in,
                              void* d_out, int out_size, void* d_ws, size_t ws_size,
                              hipStream_t stream) {
    const float* x    = (const float*)d_in[0];
    const int*   src  = (const int*)d_in[1];
    const int*   dst  = (const int*)d_in[2];
    const float* eps1 = (const float*)d_in[3];
    const float* w1a  = (const float*)d_in[4];
    const float* b1a  = (const float*)d_in[5];
    const float* g1   = (const float*)d_in[6];
    const float* bt1  = (const float*)d_in[7];
    const float* w1b  = (const float*)d_in[8];
    const float* b1b  = (const float*)d_in[9];
    const float* eps2 = (const float*)d_in[10];
    const float* w2a  = (const float*)d_in[11];
    const float* b2a  = (const float*)d_in[12];
    const float* g2   = (const float*)d_in[13];
    const float* bt2  = (const float*)d_in[14];
    const float* w2b  = (const float*)d_in[15];
    const float* b2b  = (const float*)d_in[16];

    char* ws = (char*)d_ws;
    size_t off = 0;
    auto alloc = [&](size_t bytes) -> void* {
        void* p = ws + off;
        off = (off + bytes + 255) & ~(size_t)255;
        return p;
    };
    float*  AGG   = (float*)alloc(NF * 4);
    float*  Zbuf  = (float*)alloc(NF * 4);
    float*  Ybuf  = (float*)alloc(NF * 4);
    float*  STATS = (float*)alloc(512 * 4);    // SUM | SSQ | SC | SH
    __bf16* P1a   = (__bf16*)alloc(16384 * 2);
    __bf16* P1b   = (__bf16*)alloc(16384 * 2);
    __bf16* P2a   = (__bf16*)alloc(16384 * 2);
    __bf16* P2b   = (__bf16*)alloc(16384 * 2);
    float* SUM = STATS, *SSQ = STATS + 128, *SC = STATS + 256, *SH = STATS + 384;

    const int gemm_blocks    = (NPAIR + 7) / 8;                  // 391
    const int scatter_blocks = (NEDGES * 32) / 256;              // 200000
    const int zero_blocks    = (int)((NF / 4 + 255) / 256);      // 12500

    // Pack all 4 weight matrices into WMMA B-fragment layout (bf16).
    pack_w_kernel<<<64, 256, 0, stream>>>(w1a, P1a);
    pack_w_kernel<<<64, 256, 0, stream>>>(w1b, P1b);
    pack_w_kernel<<<64, 256, 0, stream>>>(w2a, P2a);
    pack_w_kernel<<<64, 256, 0, stream>>>(w2b, P2b);

    // ---- layer 1
    zero_f4_kernel<<<zero_blocks, 256, 0, stream>>>((float4*)AGG, (int)(NF / 4));
    zero_f4_kernel<<<1, 256, 0, stream>>>((float4*)STATS, 64);   // SUM+SSQ
    scatter_kernel<<<scatter_blocks, 256, 0, stream>>>(x, src, dst, AGG);
    gemm_a_kernel<<<gemm_blocks, 256, 0, stream>>>(x, AGG, eps1, P1a, b1a, Zbuf, SUM, SSQ);
    bn_finalize_kernel<<<1, 128, 0, stream>>>(SUM, SSQ, g1, bt1, SC, SH);
    gemm_b_kernel<<<gemm_blocks, 256, 0, stream>>>(Zbuf, SC, SH, P1b, b1b, Ybuf, 1);

    // ---- layer 2
    zero_f4_kernel<<<zero_blocks, 256, 0, stream>>>((float4*)AGG, (int)(NF / 4));
    zero_f4_kernel<<<1, 256, 0, stream>>>((float4*)STATS, 64);
    scatter_kernel<<<scatter_blocks, 256, 0, stream>>>(Ybuf, src, dst, AGG);
    gemm_a_kernel<<<gemm_blocks, 256, 0, stream>>>(Ybuf, AGG, eps2, P2a, b2a, Zbuf, SUM, SSQ);
    bn_finalize_kernel<<<1, 128, 0, stream>>>(SUM, SSQ, g2, bt2, SC, SH);
    gemm_b_kernel<<<gemm_blocks, 256, 0, stream>>>(Zbuf, SC, SH, P2b, b2b, (float*)d_out, 0);
}
